// Attention_62560493633940
// MI455X (gfx1250) — matrix operand order repaired and checked by reading, not compiled, not run on hardware
//
#include <hip/hip_runtime.h>
#include <stdint.h>

// ---------------------------------------------------------------------------
// MI455X (gfx1250) implementation.  All heavy GEMMs run on
// v_wmma_f32_16x16x32_bf16 (wave32, 16x16 tiles, f32 accumulate).  Tile
// staging uses GLOBAL_LOAD_ASYNC_TO_LDS_B128 with double-buffered LDS and
// s_wait_asynccnt 4 so the async DMA of tile i+1 overlaps the WMMA work on
// tile i (async loads complete in order per ISA 4.1).
//
// Math notes (vs reference):
//  * pos_logits[h,q,k] = (p_q - p_k)·Wh[h] + bh[h].  Per softmax row (fixed
//    h,q) the p_q term and bh are constants => pos_attn[h,q,:] =
//    softmax_k(-s[k,h]) with s = p @ Wh^T.  q-independent => precompute
//    w[h,k] once (bh provably cancels).
//  * attn and pos_attn both row-sum to 1 => the final renormalization is an
//    exact no-op; gate mix fused into the attention softmax epilogue.
//  * attention scale 1/sqrt(64)=0.125 folded into RoPE'd q (bf16 store).
//  * k/v GQA repeat handled by pointer arithmetic (head h -> kv head h>>2).
//  * B rows >= N feed only discarded output columns, so the B row index is
//    clamped (valid address, garbage tolerated) instead of guarded -> the
//    staging path is divergence-free and every wave issues exactly 4 async
//    loads per tile.
// ---------------------------------------------------------------------------

typedef __attribute__((ext_vector_type(16))) __bf16 v16bf;
typedef __attribute__((ext_vector_type(8)))  float  v8f;

#define TT   1024
#define DIMX 1024
#define NH   16
#define NKV  4
#define HDX  64
#define PDX  64
#define PFX  128

__device__ __forceinline__ unsigned short f2bf(float f) {
  union { float f; unsigned u; } a; a.f = f;
  unsigned u = a.u;
  unsigned r = u + 0x7fffu + ((u >> 16) & 1u);   // round-to-nearest-even
  return (unsigned short)(r >> 16);
}

// --------------------------- elementwise kernels ---------------------------

__global__ void k_f32_to_bf16(const float* __restrict__ in,
                              unsigned short* __restrict__ out, int n) {
  int i = blockIdx.x * blockDim.x + threadIdx.x;
  if (i < n) out[i] = f2bf(in[i]);
}

// RoPE on (T, nh*64) f32 -> bf16, optional scale folded in.
// grid = T blocks, nh*32 threads.
__global__ void k_rope(const float* __restrict__ src,
                       unsigned short* __restrict__ dst, int nh, float scale) {
  int t = blockIdx.x;
  int tid = threadIdx.x;
  int h = tid >> 5, i = tid & 31;
  size_t base = (size_t)t * nh * 64 + (size_t)h * 64;
  float x1 = src[base + i];
  float x2 = src[base + 32 + i];
  // theta_i = 10000^(-2i/64)
  float theta = __expf(-(float)i * (2.0f / 64.0f) * 9.210340371976184f);
  float f = (float)t * theta;
  float c = __cosf(f), s = __sinf(f);
  dst[base + i]      = f2bf(scale * (x1 * c - x2 * s));
  dst[base + 32 + i] = f2bf(scale * (x2 * c + x1 * s));
}

// v (T x 256) f32  ->  vT (256 x T) bf16.  grid=T blocks, 256 threads.
__global__ void k_transpose_bf16(const float* __restrict__ v,
                                 unsigned short* __restrict__ vt) {
  int t = blockIdx.x, c = threadIdx.x;
  vt[(size_t)c * TT + t] = f2bf(v[(size_t)t * (NKV * HDX) + c]);
}

// w[h,k] = softmax_k(-s[k,h]).  s is (T x 16).  grid=16 blocks, 256 threads.
__global__ void k_possoft(const float* __restrict__ s, float* __restrict__ w) {
  __shared__ float red[256];
  int h = blockIdx.x, tid = threadIdx.x;
  float vals[4];
  float m = -3.4e38f;
  for (int j = 0; j < 4; ++j) {
    int k = tid + j * 256;
    vals[j] = -s[(size_t)k * NH + h];
    m = fmaxf(m, vals[j]);
  }
  red[tid] = m; __syncthreads();
  for (int st = 128; st > 0; st >>= 1) {
    if (tid < st) red[tid] = fmaxf(red[tid], red[tid + st]);
    __syncthreads();
  }
  m = red[0]; __syncthreads();
  float sum = 0.0f;
  for (int j = 0; j < 4; ++j) { vals[j] = __expf(vals[j] - m); sum += vals[j]; }
  red[tid] = sum; __syncthreads();
  for (int st = 128; st > 0; st >>= 1) {
    if (tid < st) red[tid] += red[tid + st];
    __syncthreads();
  }
  float inv = 1.0f / red[0];
  for (int j = 0; j < 4; ++j)
    w[(size_t)h * TT + tid + j * 256] = vals[j] * inv;
}

// Row softmax of S (T x T) fused with gate mix:
//   ab[q,k] = bf16( (1-g)*softmax_k(S[q,:]) + g*w[k] ),  g = sigmoid(gate[h])
// grid = T blocks, 256 threads.
__global__ void k_softmax_gate(const float* __restrict__ S,
                               const float* __restrict__ w,
                               const float* __restrict__ gate, int h,
                               unsigned short* __restrict__ ab) {
  __shared__ float red[256];
  int q = blockIdx.x, tid = threadIdx.x;
  const float* row = S + (size_t)q * TT;
  float g = 1.0f / (1.0f + __expf(-gate[h]));
  float vals[4];
  float m = -3.4e38f;
  for (int j = 0; j < 4; ++j) {
    vals[j] = row[tid + j * 256];
    m = fmaxf(m, vals[j]);
  }
  red[tid] = m; __syncthreads();
  for (int st = 128; st > 0; st >>= 1) {
    if (tid < st) red[tid] = fmaxf(red[tid], red[tid + st]);
    __syncthreads();
  }
  m = red[0]; __syncthreads();
  float sum = 0.0f;
  for (int j = 0; j < 4; ++j) { vals[j] = __expf(vals[j] - m); sum += vals[j]; }
  red[tid] = sum; __syncthreads();
  for (int st = 128; st > 0; st >>= 1) {
    if (tid < st) red[tid] += red[tid + st];
    __syncthreads();
  }
  float inv = (1.0f - g) / red[0];
  for (int j = 0; j < 4; ++j) {
    int k = tid + j * 256;
    ab[(size_t)q * TT + k] = f2bf(vals[j] * inv + g * w[k]);
  }
}

// --------------------------- WMMA GEMM (C = A * B^T) -----------------------
// A: (M x K) bf16 row-major (lda), B: (N x K) bf16 row-major (ldb).
// C written as f32 (Cf) and/or bf16 (Cb), ldc stride, optional bias+ReLU.
// Requirements used here: M % 128 == 0, K % 32 == 0, lda/ldb % 16 == 0.
// N may be arbitrary (store-guarded; B rows clamped).
//
// Tiling: 256 threads = 8 waves (4x2); block tile 128x128, BK=32; each wave
// owns a 32x64 patch = 2x4 v_wmma_f32_16x16x32_bf16 accumulators
// (8 WMMA per 12 LDS b128 fragment loads).  Double-buffered async staging:
// tile i+1 DMA (4 x global_load_async_to_lds_b128 per wave) runs under the
// tile-i WMMAs; s_wait_asynccnt 4 retires only the older batch.

#define BM 128
#define BN 128
#define BK 32
#define BKP 40   // padded LDS row stride in ushorts (80B, 16B-aligned rows)

__device__ __forceinline__ void async_b128(unsigned lds_off,
                                           const unsigned short* gptr) {
  asm volatile("global_load_async_to_lds_b128 %0, %1, off"
               :: "v"(lds_off), "v"(gptr) : "memory");
}
__device__ __forceinline__ void wait_async0() {
  asm volatile("s_wait_asynccnt 0x0" ::: "memory");
}
__device__ __forceinline__ void wait_async4() {
  asm volatile("s_wait_asynccnt 0x4" ::: "memory");
}

__global__ void __launch_bounds__(256)
k_gemm_bf16(int M, int N, int K,
            const unsigned short* __restrict__ A, int lda,
            const unsigned short* __restrict__ B, int ldb,
            float* __restrict__ Cf, unsigned short* __restrict__ Cb, int ldc,
            const float* __restrict__ bias, int relu) {
  __shared__ unsigned short As[2][BM * BKP];
  __shared__ unsigned short Bs[2][BN * BKP];

  const int tid  = threadIdx.x;
  const int lane = tid & 31;           // wave32
  const int wave = tid >> 5;           // 0..7
  const int wm   = wave >> 1;          // 0..3 : 32-row strip
  const int wn   = wave & 1;           // 0..1 : 64-col strip

  const int bm = blockIdx.x * BM;
  const int bn = blockIdx.y * BN;

  // cooperative staging: 128 rows x 32 cols per tile, 16 ushorts per thread
  const int srow = tid >> 1;           // 0..127
  const int scol = (tid & 1) * 16;     // 0 / 16 (two b128 chunks each)
  int brow = bn + srow;
  if (brow > N - 1) brow = N - 1;      // clamp: cols >= N are never stored
  const unsigned short* agp = A + (size_t)(bm + srow) * lda + scol;
  const unsigned short* bgp = B + (size_t)brow * ldb + scol;
  unsigned aL[2], bL[2];
  for (int b = 0; b < 2; ++b) {
    aL[b] = (unsigned)(uintptr_t)(&As[b][srow * BKP + scol]);
    bL[b] = (unsigned)(uintptr_t)(&Bs[b][srow * BKP + scol]);
  }

  v8f acc[2][4];
  for (int i = 0; i < 2; ++i)
    for (int j = 0; j < 4; ++j)
      acc[i][j] = (v8f){0.f, 0.f, 0.f, 0.f, 0.f, 0.f, 0.f, 0.f};

  const int r   = lane & 15;
  const int hi  = lane >> 4;           // 0: lanes 0-15, 1: lanes 16-31
  const int ak0 = hi * 8;              // A half-chunk base (ISA 16-bit A layout)
  const int bk0 = hi * 8;              // B K-range base

  // ---- async-stage a K-tile into LDS buffer `b` (4 wave-wide DMAs) ----
  auto issue = [&](int b, int k0) {
    async_b128(aL[b],      agp + k0);
    async_b128(aL[b] + 16, agp + k0 + 8);
    async_b128(bL[b],      bgp + k0);
    async_b128(bL[b] + 16, bgp + k0 + 8);
  };

  const int ntiles = K / BK;
  issue(0, 0);                                   // prologue: tile 0 in flight

  for (int t = 0; t < ntiles; ++t) {
    const int cur = t & 1;
    const bool more = (t + 1) < ntiles;
    if (more) issue(cur ^ 1, (t + 1) * BK);      // overlap DMA with WMMA below
    if (t + 2 < ntiles)                          // L2 warm two tiles ahead
      __builtin_prefetch(agp + (t + 2) * BK, 0, 3);
    if (more) wait_async4(); else wait_async0(); // retire only tile t's batch
    __syncthreads();

    // ---- build fragments per ISA wave32 VGPR layouts ----
    union { v16bf v; uint4 q[2]; } af[2], bf[4];
    for (int i = 0; i < 2; ++i) {
      const unsigned short* ap = &As[cur][(wm * 32 + i * 16 + r) * BKP];
      af[i].q[0] = *(const uint4*)(ap + ak0);        // K = ak0..ak0+7
      af[i].q[1] = *(const uint4*)(ap + ak0 + 16);   // K = ak0+16..ak0+23
    }
    for (int j = 0; j < 4; ++j) {
      const unsigned short* bp = &Bs[cur][(wn * 64 + j * 16 + r) * BKP];
      bf[j].q[0] = *(const uint4*)(bp + bk0);        // K = bk0..bk0+7
      bf[j].q[1] = *(const uint4*)(bp + bk0 + 16);   // K = bk0+16..bk0+23
    }

    // ---- 2x4 WMMA ----
    for (int i = 0; i < 2; ++i)
      for (int j = 0; j < 4; ++j)
        acc[i][j] = __builtin_amdgcn_wmma_f32_16x16x32_bf16(
            false, af[i].v, false, bf[j].v, (short)0, acc[i][j], false, false);
    __syncthreads();   // all waves done reading buffer `cur` before its reuse
  }

  // ---- epilogue: C/D layout = VGPR e -> M = e + 8*hi, N = lane&15 ----
  const int rbase = hi * 8;
  const int c16   = lane & 15;
  for (int i = 0; i < 2; ++i) {
    for (int j = 0; j < 4; ++j) {
      int gcol = bn + wn * 64 + j * 16 + c16;
      if (gcol >= N) continue;
      float bb = bias ? bias[gcol] : 0.0f;
      for (int e = 0; e < 8; ++e) {
        int grow = bm + wm * 32 + i * 16 + rbase + e;
        if (grow >= M) continue;
        float v = acc[i][j][e] + bb;
        if (relu) v = v > 0.0f ? v : 0.0f;
        if (Cf) Cf[(size_t)grow * ldc + gcol] = v;
        if (Cb) Cb[(size_t)grow * ldc + gcol] = f2bf(v);
      }
    }
  }
}

// ------------------------------- launcher ----------------------------------

extern "C" void kernel_launch(void* const* d_in, const int* in_sizes, int n_in,
                              void* d_out, int out_size, void* d_ws, size_t ws_size,
                              hipStream_t stream) {
  (void)in_sizes; (void)n_in; (void)out_size; (void)ws_size;
  const float* x    = (const float*)d_in[0];   // (1,1024,1024)
  const float* pos  = (const float*)d_in[1];   // (1,1024,64)
  const float* Wq   = (const float*)d_in[2];   // (1024,1024)
  const float* Wk   = (const float*)d_in[3];   // (256,1024)
  const float* Wv   = (const float*)d_in[4];   // (256,1024)
  const float* Wo   = (const float*)d_in[5];   // (1024,1024)
  const float* bo   = (const float*)d_in[6];   // (1024,)
  const float* Wp1  = (const float*)d_in[7];   // (64,64)
  const float* bp1  = (const float*)d_in[8];   // (64,)
  const float* Wp2  = (const float*)d_in[9];   // (128,64)
  const float* bp2  = (const float*)d_in[10];  // (128,)
  const float* Wh   = (const float*)d_in[11];  // (16,128)
  // d_in[12] = bh : cancels inside pos-softmax (constant per row) -> unused
  const float* gate = (const float*)d_in[13];  // (16,)
  float* out = (float*)d_out;

  char* wsp = (char*)d_ws;
  size_t off = 0;
  auto alloc = [&](size_t bytes) -> void* {
    void* p = wsp + off;
    off += (bytes + 255) & ~(size_t)255;
    return p;
  };
  typedef unsigned short us;
  us*    xb   = (us*)alloc((size_t)TT * DIMX * 2);
  us*    wqb  = (us*)alloc((size_t)DIMX * DIMX * 2);
  us*    wkb  = (us*)alloc((size_t)NKV * HDX * DIMX * 2);
  us*    wvb  = (us*)alloc((size_t)NKV * HDX * DIMX * 2);
  us*    wob  = (us*)alloc((size_t)DIMX * DIMX * 2);
  us*    posb = (us*)alloc((size_t)TT * PDX * 2);
  us*    wp1b = (us*)alloc((size_t)PDX * PDX * 2);
  us*    wp2b = (us*)alloc((size_t)PFX * PDX * 2);
  us*    whb  = (us*)alloc((size_t)NH * PFX * 2);
  float* qf   = (float*)alloc((size_t)TT * DIMX * 4);
  float* kf   = (float*)alloc((size_t)TT * NKV * HDX * 4);
  float* vf   = (float*)alloc((size_t)TT * NKV * HDX * 4);
  us*    qb   = (us*)alloc((size_t)TT * DIMX * 2);
  us*    kb   = (us*)alloc((size_t)TT * NKV * HDX * 2);
  us*    vtb  = (us*)alloc((size_t)NKV * HDX * TT * 2);
  us*    p1b  = (us*)alloc((size_t)TT * PDX * 2);
  us*    pb   = (us*)alloc((size_t)TT * PFX * 2);
  float* sf   = (float*)alloc((size_t)TT * NH * 4);
  float* wf   = (float*)alloc((size_t)NH * TT * 4);
  float* scf  = (float*)alloc((size_t)TT * TT * 4);
  us*    ab   = (us*)alloc((size_t)TT * TT * 2);
  us*    ctxb = (us*)alloc((size_t)TT * DIMX * 2);

  auto conv = [&](const float* src, us* dst, int n) {
    k_f32_to_bf16<<<(n + 255) / 256, 256, 0, stream>>>(src, dst, n);
  };
  auto gemm = [&](int M, int N, int K, const us* A, int lda, const us* B, int ldb,
                  float* Cf, us* Cb, int ldc, const float* bias, int relu) {
    dim3 grid(M / BM, (N + BN - 1) / BN);
    k_gemm_bf16<<<grid, 256, 0, stream>>>(M, N, K, A, lda, B, ldb,
                                          Cf, Cb, ldc, bias, relu);
  };

  // ---- bf16 conversions ----
  conv(x,   xb,   TT * DIMX);
  conv(Wq,  wqb,  DIMX * DIMX);
  conv(Wk,  wkb,  NKV * HDX * DIMX);
  conv(Wv,  wvb,  NKV * HDX * DIMX);
  conv(Wo,  wob,  DIMX * DIMX);
  conv(pos, posb, TT * PDX);
  conv(Wp1, wp1b, PDX * PDX);
  conv(Wp2, wp2b, PFX * PDX);
  conv(Wh,  whb,  NH * PFX);

  // ---- q/k/v projections ----
  gemm(TT, DIMX,      DIMX, xb, DIMX, wqb, DIMX, qf, nullptr, DIMX,      nullptr, 0);
  gemm(TT, NKV * HDX, DIMX, xb, DIMX, wkb, DIMX, kf, nullptr, NKV * HDX, nullptr, 0);
  gemm(TT, NKV * HDX, DIMX, xb, DIMX, wvb, DIMX, vf, nullptr, NKV * HDX, nullptr, 0);

  // ---- RoPE (scale 1/sqrt(64) folded into q), v transpose ----
  k_rope<<<TT, NH * 32, 0, stream>>>(qf, qb, NH, 0.125f);
  k_rope<<<TT, NKV * 32, 0, stream>>>(kf, kb, NKV, 1.0f);
  k_transpose_bf16<<<TT, NKV * HDX, 0, stream>>>(vf, vtb);

  // ---- positional path: p = relu(pos Wp1^T + bp1) Wp2^T + bp2 ; s = p Wh^T
  gemm(TT, PDX, PDX, posb, PDX, wp1b, PDX, nullptr, p1b, PDX, bp1, 1);
  gemm(TT, PFX, PDX, p1b,  PDX, wp2b, PDX, nullptr, pb,  PFX, bp2, 0);
  gemm(TT, NH,  PFX, pb,   PFX, whb,  PFX, sf, nullptr, NH, nullptr, 0);
  k_possoft<<<NH, 256, 0, stream>>>(sf, wf);

  // ---- attention, head by head (reuses score/attn buffers) ----
  for (int h = 0; h < NH; ++h) {
    const us* qh = qb + h * HDX;                         // lda = 1024
    const us* kh = kb + (h >> 2) * HDX;                  // ldb = 256 (GQA)
    gemm(TT, TT, HDX, qh, DIMX, kh, NKV * HDX, scf, nullptr, TT, nullptr, 0);
    k_softmax_gate<<<TT, 256, 0, stream>>>(scf, wf + (size_t)h * TT, gate, h, ab);
    const us* vth = vtb + (size_t)(h >> 2) * HDX * TT;   // (64 x 1024) bf16
    gemm(TT, HDX, TT, ab, TT, vth, TT, nullptr, ctxb + h * HDX, DIMX, nullptr, 0);
  }

  // ---- output projection ----
  gemm(TT, DIMX, DIMX, ctxb, DIMX, wob, DIMX, out, nullptr, DIMX, bo, 0);
}
